// LlamaAttention_38817914421649
// MI455X (gfx1250) — matrix operand order repaired and checked by reading, not compiled, not run on hardware
//
#include <hip/hip_runtime.h>

// Problem constants (match reference setup_inputs)
#define B_   2
#define S_   2048
#define H_   4096
#define NH_  32
#define DH_  128
#define M_   (B_ * S_)          // 4096 rows for all GEMMs

typedef __attribute__((ext_vector_type(16))) __bf16 v16bf;
typedef __attribute__((ext_vector_type(8)))  __bf16 v8bf;
typedef __attribute__((ext_vector_type(8)))  float  v8f;

union Frag { v16bf v; v8bf h[2]; };

__device__ __forceinline__ unsigned short f2bf(float x) {
  unsigned int u = __float_as_uint(x);
  return (unsigned short)((u + 0x7FFFu + ((u >> 16) & 1u)) >> 16); // RNE
}

// A-matrix fragment (16x32 bf16, M x K), generic/global memory, row-major ld.
__device__ __forceinline__ v16bf load_a_frag(const unsigned short* A, int lda,
                                             int mbase, int k0, int lane) {
  int m = lane & 15, half = lane >> 4;
  const unsigned short* p = A + (size_t)(mbase + m) * lda + k0 + half * 8;
  Frag f;
  f.h[0] = *(const v8bf*)(p);
  f.h[1] = *(const v8bf*)(p + 16);
  return f.v;
}

// B-matrix fragment (32x16 bf16, K x N), sourced K-contiguous from [N,K] rows.
__device__ __forceinline__ v16bf load_b_frag(const unsigned short* Wm, int ldb,
                                             int nbase, int k0, int lane) {
  int n = lane & 15, half = lane >> 4;
  const unsigned short* p = Wm + (size_t)(nbase + n) * ldb + k0 + half * 8;
  Frag f;
  f.h[0] = *(const v8bf*)(p);
  f.h[1] = *(const v8bf*)(p + 16);
  return f.v;
}

__device__ __forceinline__ v8f wmma_bf16(v16bf a, v16bf b, v8f c) {
  return __builtin_amdgcn_wmma_f32_16x16x32_bf16(false, a, false, b,
                                                 (short)0, c, false, false);
}

// Async copy 16B global -> LDS (ASYNCcnt-tracked). lds_off = raw LDS byte addr.
__device__ __forceinline__ void async_copy_b128(unsigned int lds_off,
                                                const void* gaddr) {
  asm volatile("global_load_async_to_lds_b128 %0, %1, off"
               :: "v"(lds_off), "v"((unsigned long long)(uintptr_t)gaddr)
               : "memory");
}

__device__ __forceinline__ void wait_async_all() {
  asm volatile("s_wait_asynccnt 0x0" ::: "memory");
}

// ---------------------------------------------------------------- cast f32->bf16
__global__ void cast_bf16_kernel(const float* __restrict__ in,
                                 unsigned short* __restrict__ out, int n) {
  int i = blockIdx.x * blockDim.x + threadIdx.x;
  int stride = gridDim.x * blockDim.x;
  for (; i < n; i += stride) out[i] = f2bf(in[i]);
}

// ---------------------------------------------------------------- GEMM: C = A * W^T
// A: [M,K] bf16. W: [N,K] bf16. C: [M,N] f32.
// Block tile 128x64, 8 waves of 32x32. Double-buffered async-LDS staging.
#define LDT 40   // padded LDS row stride (bf16 elems) for a 32-wide K slab

__global__ __launch_bounds__(256) void gemm_bf16_lds_kernel(
    const unsigned short* __restrict__ A, const unsigned short* __restrict__ W,
    float* __restrict__ C, int M, int N, int K) {
  __shared__ __align__(16) unsigned short Abuf[2][128 * LDT];
  __shared__ __align__(16) unsigned short Bbuf[2][64 * LDT];

  int tid = threadIdx.x;
  int lane = tid & 31, wave = tid >> 5;
  int mBlocks = M >> 7;
  int bm = blockIdx.x % mBlocks;
  int bn = blockIdx.x / mBlocks;
  int blockM = bm << 7, blockN = bn << 6;
  int wm = wave >> 1, wn = wave & 1;          // 4x2 wave grid of 32x32 tiles

  unsigned int aBase[2], bBase[2];
  aBase[0] = (unsigned int)(uintptr_t)&Abuf[0][0];
  aBase[1] = (unsigned int)(uintptr_t)&Abuf[1][0];
  bBase[0] = (unsigned int)(uintptr_t)&Bbuf[0][0];
  bBase[1] = (unsigned int)(uintptr_t)&Bbuf[1][0];

  // One K-slab = 128x32 A (512 x b128) + 64x32 B (256 x b128): 3 copies/thread.
  auto issue_slab = [&](int buf, int k0) {
#pragma unroll
    for (int j = 0; j < 2; ++j) {
      int c = tid + j * 256;
      int row = c >> 2, seg = c & 3;
      const unsigned short* g = A + (size_t)(blockM + row) * K + k0 + seg * 8;
      async_copy_b128(aBase[buf] + (unsigned int)(row * LDT + seg * 8) * 2, g);
    }
    {
      int row = tid >> 2, seg = tid & 3;
      const unsigned short* g = W + (size_t)(blockN + row) * K + k0 + seg * 8;
      async_copy_b128(bBase[buf] + (unsigned int)(row * LDT + seg * 8) * 2, g);
    }
  };

  v8f acc[2][2] = {{v8f{}, v8f{}}, {v8f{}, v8f{}}};

  issue_slab(0, 0);
  int nK = K >> 5;
  for (int i = 0; i < nK; ++i) {
    int cur = i & 1;
    wait_async_all();
    __syncthreads();                          // slab `cur` ready everywhere
    if (i + 1 < nK) issue_slab(1 - cur, (i + 1) << 5);

    const unsigned short* As = &Abuf[cur][0];
    const unsigned short* Bs = &Bbuf[cur][0];
    v16bf a0 = load_a_frag(As, LDT, wm * 32, 0, lane);
    v16bf a1 = load_a_frag(As, LDT, wm * 32 + 16, 0, lane);
    v16bf b0 = load_b_frag(Bs, LDT, wn * 32, 0, lane);
    v16bf b1 = load_b_frag(Bs, LDT, wn * 32 + 16, 0, lane);
    acc[0][0] = wmma_bf16(a0, b0, acc[0][0]);
    acc[0][1] = wmma_bf16(a0, b1, acc[0][1]);
    acc[1][0] = wmma_bf16(a1, b0, acc[1][0]);
    acc[1][1] = wmma_bf16(a1, b1, acc[1][1]);
  }

  int half = lane >> 4, col = lane & 15;
#pragma unroll
  for (int mi = 0; mi < 2; ++mi) {
#pragma unroll
    for (int ni = 0; ni < 2; ++ni) {
      float* out = C + (size_t)(blockM + wm * 32 + mi * 16) * N +
                   blockN + wn * 32 + ni * 16 + col;
#pragma unroll
      for (int r = 0; r < 8; ++r)
        out[(size_t)(r + 8 * half) * N] = acc[mi][ni][r];
    }
  }
}

// ---------------------------------------------------------------- RoPE + repack
// in: [B,S,H] f32 (proj output).  out: [B, NH, S, DH] bf16 with RoPE applied.
__global__ void rope_pack_kernel(const float* __restrict__ in,
                                 unsigned short* __restrict__ out) {
  int n_total = B_ * S_ * H_;
  int i = blockIdx.x * blockDim.x + threadIdx.x;
  int stride = gridDim.x * blockDim.x;
  for (; i < n_total; i += stride) {
    int n = i & (H_ - 1);
    int bs = i >> 12;              // / H_
    int s = bs & (S_ - 1);
    int b = bs >> 11;              // / S_
    int d = n & (DH_ - 1);
    int h = n >> 7;                // / DH_
    int j = d & 63;
    float inv = __expf(-(float)j * (1.0f / 64.0f) * 9.2103403719761836f);
    float ang = (float)s * inv;
    float sn, cs;
    __sincosf(ang, &sn, &cs);
    float x = in[i];
    float other = (d < 64) ? -in[i + 64] : in[i - 64];
    out[(((size_t)(b * NH_ + h) * S_ + s) * DH_) + d] = f2bf(x * cs + other * sn);
  }
}

// V repack transposed: in [B,S,H] f32 -> out [B, NH, DH, S] bf16.
__global__ void vpack_kernel(const float* __restrict__ in,
                             unsigned short* __restrict__ out) {
  int n_total = B_ * S_ * H_;
  int i = blockIdx.x * blockDim.x + threadIdx.x;
  int stride = gridDim.x * blockDim.x;
  for (; i < n_total; i += stride) {
    int n = i & (H_ - 1);
    int bs = i >> 12;
    int s = bs & (S_ - 1);
    int b = bs >> 11;
    int d = n & (DH_ - 1);
    int h = n >> 7;
    out[((size_t)((b * NH_ + h) * DH_ + d)) * S_ + s] = f2bf(in[i]);
  }
}

// ---------------------------------------------------------------- flash attention
// One wave per (b, h, 16-row q tile). Online softmax, causal mask analytic.
__global__ __launch_bounds__(256) void attn_kernel(
    const unsigned short* __restrict__ Qb, const unsigned short* __restrict__ Kb,
    const unsigned short* __restrict__ Vt, unsigned short* __restrict__ ctx) {
  __shared__ unsigned short pls[8][16 * 32];   // per-wave P staging (bf16)

  int lane = threadIdx.x & 31;
  int wave = threadIdx.x >> 5;
  int gwave = blockIdx.x * 8 + wave;           // 0 .. B*NH*(S/16)-1
  int qt = gwave & 127;                        // S/16 = 128
  int bh = gwave >> 7;
  int b = bh >> 5, h = bh & 31;

  const unsigned short* Q = Qb + (size_t)bh * S_ * DH_;
  const unsigned short* Kp = Kb + (size_t)bh * S_ * DH_;
  const unsigned short* Vp = Vt + (size_t)bh * DH_ * S_;

  int col = lane & 15, half = lane >> 4;
  const float scale = 0.08838834764831845f;    // 1/sqrt(128)

  v16bf qf[4];
#pragma unroll
  for (int j = 0; j < 4; ++j) qf[j] = load_a_frag(Q, DH_, qt * 16, j * 32, lane);

  v8f o[8] = {v8f{}, v8f{}, v8f{}, v8f{}, v8f{}, v8f{}, v8f{}, v8f{}};
  float mrow[8], lrow[8];
#pragma unroll
  for (int r = 0; r < 8; ++r) { mrow[r] = -1e30f; lrow[r] = 0.0f; }

  int kvEnd = qt * 16 + 16;
  for (int c0 = 0; c0 < kvEnd; c0 += 32) {
    if (c0 + 32 < kvEnd) {                     // hint next KV chunk into cache
      __builtin_prefetch(Kp + (size_t)(c0 + 32) * DH_, 0, 0);
      __builtin_prefetch(Vp + (size_t)col * S_ + c0 + 32, 0, 0);
    }
    v8f s0 = v8f{}, s1 = v8f{};
#pragma unroll
    for (int j = 0; j < 4; ++j) {
      v16bf bk = load_b_frag(Kp, DH_, c0, j * 32, lane);
      s0 = wmma_bf16(qf[j], bk, s0);
    }
#pragma unroll
    for (int j = 0; j < 4; ++j) {
      v16bf bk = load_b_frag(Kp, DH_, c0 + 16, j * 32, lane);
      s1 = wmma_bf16(qf[j], bk, s1);
    }

#pragma unroll
    for (int r = 0; r < 8; ++r) {
      int rq = qt * 16 + r + 8 * half;
      float v0 = s0[r] * scale;
      float v1 = s1[r] * scale;
      if (c0 + col > rq)      v0 = -1e30f;
      if (c0 + 16 + col > rq) v1 = -1e30f;
      float tm = fmaxf(v0, v1);
#pragma unroll
      for (int off = 8; off > 0; off >>= 1) tm = fmaxf(tm, __shfl_xor(tm, off, 16));
      float mnew = fmaxf(mrow[r], tm);
      float alpha = __expf(mrow[r] - mnew);
      float p0 = __expf(v0 - mnew);
      float p1 = __expf(v1 - mnew);
      float rs = p0 + p1;
#pragma unroll
      for (int off = 8; off > 0; off >>= 1) rs += __shfl_xor(rs, off, 16);
      lrow[r] = lrow[r] * alpha + rs;
      mrow[r] = mnew;
#pragma unroll
      for (int t = 0; t < 8; ++t) o[t][r] *= alpha;
      int rowm = r + 8 * half;
      pls[wave][rowm * 32 + col]      = f2bf(p0);
      pls[wave][rowm * 32 + 16 + col] = f2bf(p1);
    }
    asm volatile("s_wait_dscnt 0x0" ::: "memory");

    Frag pf;                                   // D-layout -> A-layout via LDS
    const unsigned short* pl = &pls[wave][col * 32] + half * 8;
    pf.h[0] = *(const v8bf*)(pl);
    pf.h[1] = *(const v8bf*)(pl + 16);

#pragma unroll
    for (int t = 0; t < 8; ++t) {
      const unsigned short* vb = Vp + (size_t)(t * 16 + col) * S_ + c0 + half * 8;
      Frag bfrag;
      bfrag.h[0] = *(const v8bf*)(vb);
      bfrag.h[1] = *(const v8bf*)(vb + 16);
      o[t] = wmma_bf16(pf.v, bfrag.v, o[t]);
    }
  }

#pragma unroll
  for (int t = 0; t < 8; ++t) {
#pragma unroll
    for (int r = 0; r < 8; ++r) {
      int rq = qt * 16 + r + 8 * half;
      float val = o[t][r] / lrow[r];
      ctx[((size_t)(b * S_ + rq)) * H_ + h * DH_ + t * 16 + col] = f2bf(val);
    }
  }
}

// ---------------------------------------------------------------- launch
extern "C" void kernel_launch(void* const* d_in, const int* in_sizes, int n_in,
                              void* d_out, int out_size, void* d_ws, size_t ws_size,
                              hipStream_t stream) {
  const float* hidden = (const float*)d_in[0];
  const float* Wq = (const float*)d_in[1];
  const float* Wk = (const float*)d_in[2];
  const float* Wv = (const float*)d_in[3];
  const float* Wo = (const float*)d_in[4];
  // d_in[5] attn_mask (causal, applied analytically), d_in[6] position_ids (== s)

  const size_t nBSH = (size_t)B_ * S_ * H_;
  const size_t nHH  = (size_t)H_ * H_;

  unsigned short* xb  = (unsigned short*)d_ws;
  unsigned short* wqb = xb + nBSH;
  unsigned short* wkb = wqb + nHH;
  unsigned short* wvb = wkb + nHH;
  unsigned short* wob = wvb + nHH;
  unsigned short* qb  = wob + nHH;
  unsigned short* kb  = qb + nBSH;
  unsigned short* vt  = kb + nBSH;
  unsigned short* ctx = vt + nBSH;
  float* tmp = (float*)(ctx + nBSH);
  size_t needed = (5 * nBSH + 4 * nHH) * 2 + nBSH * 4;
  if (ws_size < needed) return;

  dim3 blk(256);
  cast_bf16_kernel<<<4096, blk, 0, stream>>>(hidden, xb, (int)nBSH);
  cast_bf16_kernel<<<4096, blk, 0, stream>>>(Wq, wqb, (int)nHH);
  cast_bf16_kernel<<<4096, blk, 0, stream>>>(Wk, wkb, (int)nHH);
  cast_bf16_kernel<<<4096, blk, 0, stream>>>(Wv, wvb, (int)nHH);
  cast_bf16_kernel<<<4096, blk, 0, stream>>>(Wo, wob, (int)nHH);

  // blocks = (M/128)*(N/64) = 32*64 = 2048
  const int gemmBlocks = (M_ / 128) * (H_ / 64);

  gemm_bf16_lds_kernel<<<gemmBlocks, blk, 0, stream>>>(xb, wqb, tmp, M_, H_, H_);
  rope_pack_kernel<<<4096, blk, 0, stream>>>(tmp, qb);
  gemm_bf16_lds_kernel<<<gemmBlocks, blk, 0, stream>>>(xb, wkb, tmp, M_, H_, H_);
  rope_pack_kernel<<<4096, blk, 0, stream>>>(tmp, kb);
  gemm_bf16_lds_kernel<<<gemmBlocks, blk, 0, stream>>>(xb, wvb, tmp, M_, H_, H_);
  vpack_kernel<<<4096, blk, 0, stream>>>(tmp, vt);

  attn_kernel<<<1024, blk, 0, stream>>>(qb, kb, vt, ctx);

  gemm_bf16_lds_kernel<<<gemmBlocks, blk, 0, stream>>>(ctx, wob, (float*)d_out, M_, H_, H_);
}
